// DynamicNet_58342835749293
// MI455X (gfx1250) — compile-verified, matrix operand order checked by reading, not run on hardware
//
#include <hip/hip_runtime.h>
#include <hip/hip_bf16.h>
#include <math.h>

typedef __attribute__((ext_vector_type(16))) _Float16 v16h;
typedef __attribute__((ext_vector_type(8)))  float    v8f;

static inline unsigned cdiv(size_t a, size_t b) { return (unsigned)((a + b - 1) / b); }

// WMMA 16x16x32 f16 fragment addressing: for K-column kk (0..31), the owning
// half-wave is hi=(kk>>3)&1 (lane += 16*hi) and packed element e within the
// 16-half per-lane fragment is e = (kk&7) | ((kk&16)>>1).
__device__ __forceinline__ int frag_hi(int kk) { return (kk >> 3) & 1; }
__device__ __forceinline__ int frag_e(int kk) { return (kk & 7) | ((kk & 16) >> 1); }

// ---------------------------------------------------------------------------
// Utility kernels
// ---------------------------------------------------------------------------
__global__ void zero_kernel(float* p, size_t n) {
    size_t i = (size_t)blockIdx.x * blockDim.x + threadIdx.x;
    if (i < n) p[i] = 0.0f;
}

// in: [B,H,W,3] -> out: [B,3,H,W]
__global__ void hwc2chw_kernel(const float* __restrict__ in, float* __restrict__ out,
                               int H, int W, size_t n) {
    size_t g = (size_t)blockIdx.x * blockDim.x + threadIdx.x;
    if (g >= n) return;
    size_t t = g;
    int j = (int)(t % W); t /= W;
    int i = (int)(t % H); t /= H;
    int c = (int)(t % 3); t /= 3;
    int b = (int)t;
    out[g] = in[(((size_t)b * H + i) * W + j) * 3 + c];
}

// feat: [B,C,L] -> X: [B*L, C]
__global__ void chw2rows_kernel(const float* __restrict__ fin, float* __restrict__ X,
                                int C, int L, size_t n) {
    size_t g = (size_t)blockIdx.x * blockDim.x + threadIdx.x;
    if (g >= n) return;
    size_t t = g;
    int c = (int)(t % C); t /= C;
    int l = (int)(t % L); t /= L;
    int b = (int)t;
    X[g] = fin[((size_t)b * C + c) * L + l];
}

// ---------------------------------------------------------------------------
// Depth transform: 27 -> 64 (BN over (B,L), relu) -> 9
// ---------------------------------------------------------------------------
__global__ __launch_bounds__(256)
void dt1_kernel(const float* __restrict__ xyz, const float* __restrict__ w1,
                const float* __restrict__ b1, float* __restrict__ hbuf,
                float* __restrict__ stats, int H, int W, int pad, int dil, size_t n) {
    __shared__ float ssum[64], ssq[64];
    int tid = threadIdx.x;
    if (tid < 64) { ssum[tid] = 0.0f; ssq[tid] = 0.0f; }
    __syncthreads();
    int L = H * W;
    size_t g = (size_t)blockIdx.x * blockDim.x + tid;
    if (g < n) {
        int b = (int)(g / L), l = (int)(g % L);
        int i = l / W, j = l % W;
        float patch[27];
        for (int c = 0; c < 3; ++c)
            for (int p = 0; p < 9; ++p) {
                int si = i + (p / 3) * dil - pad;
                int sj = j + (p % 3) * dil - pad;
                float v = 0.0f;
                if (si >= 0 && si < H && sj >= 0 && sj < W)
                    v = xyz[((size_t)b * 3 + c) * L + si * W + sj];
                patch[c * 9 + p] = v;
            }
        float* hrow = hbuf + g * 64;
        for (int f = 0; f < 64; ++f) {
            float acc = b1[f];
#pragma unroll
            for (int q = 0; q < 27; ++q) acc += w1[f * 27 + q] * patch[q];
            hrow[f] = acc;
            atomicAdd(&ssum[f], acc);
            atomicAdd(&ssq[f], acc * acc);
        }
    }
    __syncthreads();
    if (tid < 64) {
        atomicAdd(&stats[tid], ssum[tid]);
        atomicAdd(&stats[64 + tid], ssq[tid]);
    }
}

__global__ void dt_fin_kernel(float* stats, float cnt) {
    int f = threadIdx.x;
    if (f >= 64) return;
    float mean = stats[f] / cnt;
    float var = stats[64 + f] / cnt - mean * mean;
    stats[f] = mean;
    stats[64 + f] = rsqrtf(var + 1e-5f);
}

__global__ __launch_bounds__(256)
void dt2_kernel(const float* __restrict__ hbuf, const float* __restrict__ stats,
                const float* __restrict__ gamma, const float* __restrict__ beta,
                const float* __restrict__ w2, const float* __restrict__ b2,
                float* __restrict__ tmat, int L, size_t n) {
    size_t g = (size_t)blockIdx.x * blockDim.x + threadIdx.x;
    if (g >= n) return;
    int b = (int)(g / L), l = (int)(g % L);
    const float* hrow = hbuf + g * 64;
    float hn[64];
    for (int f = 0; f < 64; ++f) {
        float v = (hrow[f] - stats[f]) * stats[64 + f] * gamma[f] + beta[f];
        hn[f] = v > 0.0f ? v : 0.0f;
    }
    for (int r = 0; r < 9; ++r) {
        float acc = b2[r];
        for (int f = 0; f < 64; ++f) acc += w2[r * 64 + f] * hn[f];
        tmat[((size_t)b * 9 + r) * L + l] = acc;
    }
}

// ---------------------------------------------------------------------------
// Dynamic conv GEMM via WMMA:  G[b][l][o] = sum_k patch(l,k)*tmat(p,l) * W[o*K+k]
// Block = 128 threads (4 waves). Wave w covers N-subtile [w*16, w*16+16).
// K unrolled by 64: two fragment-staged WMMAs per iteration, 2 accumulators.
// ---------------------------------------------------------------------------
__global__ __launch_bounds__(128)
void dynconv_wmma_kernel(const float* __restrict__ feat, const float* __restrict__ tmat,
                         const float* __restrict__ w, float* __restrict__ G,
                         int C, int H, int W, int O, int pad, int dil) {
    const int K = C * 9;
    const int L = H * W;
    __shared__ v16h AsF[2][32];      // [k-half][lane] A fragments (fragment order)
    __shared__ v16h BsF[2][4][32];   // [k-half][wave-tile][lane] B fragments
    int b = blockIdx.z;
    int l0 = blockIdx.x * 16;
    int o0 = blockIdx.y * 64;
    int tid = threadIdx.x;
    int wave = tid >> 5;
    int lane = tid & 31;
    const float* featb = feat + (size_t)b * C * L;
    const float* tmatb = tmat + (size_t)b * 9 * L;
    v8f acc0 = {}, acc1 = {};
    for (int k0 = 0; k0 < K; k0 += 64) {
        __syncthreads();
        // A tiles: 2 x (16 pixels x 32 k), written directly in fragment order
        for (int e = tid; e < 2 * 16 * 32; e += 128) {
            int hf = e >> 9;
            int r = (e >> 5) & 15;
            int kk = e & 31;
            int l = l0 + r, k = k0 + hf * 32 + kk;
            float v = 0.0f;
            if (l < L && k < K) {
                int c = k / 9, p = k % 9;
                int i = l / W, j = l % W;
                int si = i + (p / 3) * dil - pad;
                int sj = j + (p % 3) * dil - pad;
                float fv = 0.0f;
                if (si >= 0 && si < H && sj >= 0 && sj < W)
                    fv = featb[(size_t)c * L + si * W + sj];
                v = fv * tmatb[(size_t)p * L + l];
            }
            ((_Float16*)&AsF[hf][r + frag_hi(kk) * 16])[frag_e(kk)] = (_Float16)v;
        }
        // B tiles: 2 x (32 k x 64 out-channels), fragment order
        for (int e = tid; e < 2 * 32 * 64; e += 128) {
            int hf = e >> 11;
            int kk = (e >> 6) & 31;
            int oo = e & 63;
            int k = k0 + hf * 32 + kk, o = o0 + oo;
            float v = 0.0f;
            if (k < K && o < O) v = w[(size_t)o * K + k];
            ((_Float16*)&BsF[hf][oo >> 4][(oo & 15) + frag_hi(kk) * 16])[frag_e(kk)] =
                (_Float16)v;
        }
        // prefetch next weight slice into cache hierarchy
        if (k0 + 64 < K && o0 + (tid & 63) < O)
            __builtin_prefetch(w + (size_t)(o0 + (tid & 63)) * K + k0 + 64, 0, 1);
        __syncthreads();
        v16h a0 = AsF[0][lane];
        v16h b0 = BsF[0][wave][lane];
        v16h a1 = AsF[1][lane];
        v16h b1 = BsF[1][wave][lane];
        acc0 = __builtin_amdgcn_wmma_f32_16x16x32_f16(false, a0, false, b0,
                                                      (short)0, acc0, false, false);
        acc1 = __builtin_amdgcn_wmma_f32_16x16x32_f16(false, a1, false, b1,
                                                      (short)0, acc1, false, false);
    }
    v8f acc = acc0 + acc1;
    int nn = o0 + wave * 16 + (lane & 15);
    int hi = lane >> 4;
    if (nn < O) {
#pragma unroll
        for (int r = 0; r < 8; ++r) {
            int l = l0 + hi * 8 + r;
            if (l < L) G[((size_t)b * L + l) * O + nn] = acc[r];
        }
    }
}

// Faithful torch .view reinterpret: F[b][o][l] = relu(G_flat[b][o*L + l] + bias[o])
__global__ void reinterp_kernel(const float* __restrict__ G, const float* __restrict__ bias,
                                float* __restrict__ Fo, int O, int L, size_t n) {
    size_t g = (size_t)blockIdx.x * blockDim.x + threadIdx.x;
    if (g >= n) return;
    size_t t = g;
    int l = (int)(t % L); t /= L;
    int o = (int)(t % O); t /= O;
    int b = (int)t;
    float v = G[(size_t)b * L * O + (size_t)o * L + l] + bias[o];
    Fo[g] = v > 0.0f ? v : 0.0f;
}

// ---------------------------------------------------------------------------
// MaxPool(3,2,1) + unpool grid scatter + xyz pooling
// ---------------------------------------------------------------------------
__global__ void mp1_kernel(const float* __restrict__ fin, float* __restrict__ pooled,
                           float* __restrict__ grid, int C, int H, int W,
                           int OH, int OW, size_t n) {
    size_t g = (size_t)blockIdx.x * blockDim.x + threadIdx.x;
    if (g >= n) return;
    size_t t = g;
    int ow = (int)(t % OW); t /= OW;
    int oh = (int)(t % OH); t /= OH;
    int c = (int)(t % C); t /= C;
    int b = (int)t;
    const float* f = fin + ((size_t)b * C + c) * H * W;
    float best = -INFINITY;
    int arg = 0;
    for (int p = 0; p < 9; ++p) {
        int si = 2 * oh - 1 + p / 3, sj = 2 * ow - 1 + p % 3;
        float v = (si >= 0 && si < H && sj >= 0 && sj < W) ? f[si * W + sj] : -INFINITY;
        if (v > best) { best = v; arg = p; }
    }
    pooled[g] = best;
    int gi = 2 * oh - 1 + arg / 3, gj = 2 * ow - 1 + arg % 3;
    if (best != 0.0f && gi >= 0 && gi < H && gj >= 0 && gj < W)
        grid[((size_t)b * C + c) * H * W + gi * W + gj] = 1.0f;
}

__global__ void mp2_kernel(const float* __restrict__ grid, float* __restrict__ gs,
                           int C, int H, int W, size_t n) {
    size_t g = (size_t)blockIdx.x * blockDim.x + threadIdx.x;
    if (g >= n) return;
    int pix = (int)(g % ((size_t)H * W));
    int b = (int)(g / ((size_t)H * W));
    float s = 0.0f;
    for (int c = 0; c < C; ++c) s += grid[((size_t)b * C + c) * H * W + pix];
    gs[g] = s / (float)C;
}

__global__ void mp3_kernel(const float* __restrict__ gs, const float* __restrict__ xin,
                           float* __restrict__ xout, int H, int W, int OH, int OW,
                           size_t n) {
    size_t g = (size_t)blockIdx.x * blockDim.x + threadIdx.x;
    if (g >= n) return;
    size_t t = g;
    int ow = (int)(t % OW); t /= OW;
    int oh = (int)(t % OH); t /= OH;
    int b = (int)t;
    float wgt[9];
    for (int p = 0; p < 9; ++p) {
        int si = 2 * oh - 1 + p / 3, sj = 2 * ow - 1 + p % 3;
        wgt[p] = (si >= 0 && si < H && sj >= 0 && sj < W)
                     ? gs[(size_t)b * H * W + si * W + sj] : 0.0f;
    }
    for (int ch = 0; ch < 3; ++ch) {
        float s = 0.0f;
        for (int p = 0; p < 9; ++p) {
            int si = 2 * oh - 1 + p / 3, sj = 2 * ow - 1 + p % 3;
            float xv = (si >= 0 && si < H && sj >= 0 && sj < W)
                           ? xin[((size_t)b * 3 + ch) * H * W + si * W + sj] : 0.0f;
            s += xv * wgt[p];
        }
        xout[((size_t)b * 3 + ch) * OH * OW + oh * OW + ow] = s;
    }
}

// pool5 (mode 0: max, -inf pad) / pool5a (mode 1: avg, zero pad, /9), 3x3 s1 p1
__global__ void pool_s1_kernel(const float* __restrict__ fin, float* __restrict__ fout,
                               int C, int H, int W, int mode, size_t n) {
    size_t g = (size_t)blockIdx.x * blockDim.x + threadIdx.x;
    if (g >= n) return;
    size_t t = g;
    int j = (int)(t % W); t /= W;
    int i = (int)(t % H); t /= H;
    int c = (int)(t % C); t /= C;
    int b = (int)t;
    const float* f = fin + ((size_t)b * C + c) * H * W;
    if (mode == 0) {
        float best = -INFINITY;
        for (int p = 0; p < 9; ++p) {
            int si = i - 1 + p / 3, sj = j - 1 + p % 3;
            float v = (si >= 0 && si < H && sj >= 0 && sj < W) ? f[si * W + sj] : -INFINITY;
            best = v > best ? v : best;
        }
        fout[g] = best;
    } else {
        float s = 0.0f;
        for (int p = 0; p < 9; ++p) {
            int si = i - 1 + p / 3, sj = j - 1 + p % 3;
            if (si >= 0 && si < H && sj >= 0 && sj < W) s += f[si * W + sj];
        }
        fout[g] = s * (1.0f / 9.0f);
    }
}

// ---------------------------------------------------------------------------
// Generic WMMA GEMM: G[m][n] = act(sum_k X[m][k]*Wt[n][k] + bias[n])
// Same fragment-staged, K-by-64 double-accumulator scheme.
// ---------------------------------------------------------------------------
__global__ __launch_bounds__(128)
void gemm_xwT_wmma_kernel(const float* __restrict__ X, const float* __restrict__ Wt,
                          const float* __restrict__ bias, float* __restrict__ G,
                          int M, int K, int N, int relu) {
    __shared__ v16h AsF[2][32];
    __shared__ v16h BsF[2][4][32];
    int m0 = blockIdx.x * 16;
    int n0 = blockIdx.y * 64;
    int tid = threadIdx.x;
    int wave = tid >> 5;
    int lane = tid & 31;
    v8f acc0 = {}, acc1 = {};
    for (int k0 = 0; k0 < K; k0 += 64) {
        __syncthreads();
        for (int e = tid; e < 2 * 16 * 32; e += 128) {
            int hf = e >> 9;
            int r = (e >> 5) & 15;
            int kk = e & 31;
            int m = m0 + r, k = k0 + hf * 32 + kk;
            float v = (m < M && k < K) ? X[(size_t)m * K + k] : 0.0f;
            ((_Float16*)&AsF[hf][r + frag_hi(kk) * 16])[frag_e(kk)] = (_Float16)v;
        }
        for (int e = tid; e < 2 * 32 * 64; e += 128) {
            int hf = e >> 11;
            int kk = (e >> 6) & 31;
            int nn = e & 63;
            int k = k0 + hf * 32 + kk, nc = n0 + nn;
            float v = (k < K && nc < N) ? Wt[(size_t)nc * K + k] : 0.0f;
            ((_Float16*)&BsF[hf][nn >> 4][(nn & 15) + frag_hi(kk) * 16])[frag_e(kk)] =
                (_Float16)v;
        }
        if (k0 + 64 < K && n0 + (tid & 63) < N)
            __builtin_prefetch(Wt + (size_t)(n0 + (tid & 63)) * K + k0 + 64, 0, 1);
        __syncthreads();
        v16h a0 = AsF[0][lane];
        v16h b0 = BsF[0][wave][lane];
        v16h a1 = AsF[1][lane];
        v16h b1 = BsF[1][wave][lane];
        acc0 = __builtin_amdgcn_wmma_f32_16x16x32_f16(false, a0, false, b0,
                                                      (short)0, acc0, false, false);
        acc1 = __builtin_amdgcn_wmma_f32_16x16x32_f16(false, a1, false, b1,
                                                      (short)0, acc1, false, false);
    }
    v8f acc = acc0 + acc1;
    int nn = n0 + wave * 16 + (lane & 15);
    int hi = lane >> 4;
    if (nn < N) {
        float bb = bias ? bias[nn] : 0.0f;
#pragma unroll
        for (int r = 0; r < 8; ++r) {
            int m = m0 + hi * 8 + r;
            if (m < M) {
                float v = acc[r] + bb;
                if (relu) v = v > 0.0f ? v : 0.0f;
                G[(size_t)m * N + nn] = v;
            }
        }
    }
}

// ---------------------------------------------------------------------------
// Host orchestration
// ---------------------------------------------------------------------------
extern "C" void kernel_launch(void* const* d_in, const int* in_sizes, int n_in,
                              void* d_out, int out_size, void* d_ws, size_t ws_size,
                              hipStream_t stream) {
    (void)in_sizes; (void)n_in; (void)out_size; (void)ws_size;
    const float* rgb = (const float*)d_in[0];
    const float* xyz = (const float*)d_in[1];
    auto P = [&](int i) { return (const float*)d_in[i]; };

    const int B = 2;
    const size_t NF  = (size_t)B * 64 * 224 * 224;  // 6,422,528 floats
    const size_t TMF = (size_t)B * 9 * 50176;
    const size_t GSF = (size_t)B * 224 * 224;
    const size_t XYF = (size_t)B * 3 * 50176;
    float* ws = (float*)d_ws;
    float* F1 = ws;
    float* F2 = F1 + NF;
    float* SM = F2 + NF;   // shared scratch: MLP hidden / GEMM out / unpool grid / FC bufs
    float* TM = SM + NF;
    float* GS = TM + TMF;
    float* XA = GS + GSF;
    float* XB = XA + XYF;
    float* ST = XB + XYF;  // 128 floats of BN stats

    auto zero = [&](float* p, size_t n) {
        zero_kernel<<<cdiv(n, 256), 256, 0, stream>>>(p, n);
    };

    // layout transposes: [B,H,W,3] -> [B,3,H,W]
    {
        size_t n = (size_t)B * 224 * 224 * 3;
        hwc2chw_kernel<<<cdiv(n, 256), 256, 0, stream>>>(rgb, F1, 224, 224, n);
        hwc2chw_kernel<<<cdiv(n, 256), 256, 0, stream>>>(xyz, XA, 224, 224, n);
    }

    auto run_dt = [&](const float* xp, int H, int W, int pad, int dil, int pi) {
        int L = H * W;
        size_t n = (size_t)B * L;
        zero(ST, 128);
        dt1_kernel<<<cdiv(n, 256), 256, 0, stream>>>(xp, P(pi), P(pi + 1), SM, ST,
                                                     H, W, pad, dil, n);
        dt_fin_kernel<<<1, 64, 0, stream>>>(ST, (float)n);
        dt2_kernel<<<cdiv(n, 256), 256, 0, stream>>>(SM, ST, P(pi + 2), P(pi + 3),
                                                     P(pi + 4), P(pi + 5), TM, L, n);
    };

    auto run_conv = [&](const float* fin, int C, int wi, int O, int H, int W,
                        int pad, int dil, float* fout) {
        int L = H * W;
        dim3 g(cdiv(L, 16), cdiv(O, 64), B);
        dynconv_wmma_kernel<<<g, 128, 0, stream>>>(fin, TM, P(wi), SM, C, H, W, O, pad, dil);
        size_t n = (size_t)B * O * L;
        reinterp_kernel<<<cdiv(n, 256), 256, 0, stream>>>(SM, P(wi + 1), fout, O, L, n);
    };

    auto run_pool = [&](const float* fin, int C, int H, int W, const float* xin,
                        float* fout, float* xout) {
        int OH = (H + 2 - 3) / 2 + 1, OW = (W + 2 - 3) / 2 + 1;
        zero(SM, (size_t)B * C * H * W);
        size_t n1 = (size_t)B * C * OH * OW;
        mp1_kernel<<<cdiv(n1, 256), 256, 0, stream>>>(fin, fout, SM, C, H, W, OH, OW, n1);
        size_t n2 = (size_t)B * H * W;
        mp2_kernel<<<cdiv(n2, 256), 256, 0, stream>>>(SM, GS, C, H, W, n2);
        size_t n3 = (size_t)B * OH * OW;
        mp3_kernel<<<cdiv(n3, 256), 256, 0, stream>>>(GS, xin, xout, H, W, OH, OW, n3);
    };

    // Level 1 @224
    run_dt(XA, 224, 224, 1, 1, 2);
    run_conv(F1, 3, 8, 64, 224, 224, 1, 1, F2);
    run_conv(F2, 64, 10, 64, 224, 224, 1, 1, F1);
    run_pool(F1, 64, 224, 224, XA, F2, XB);
    // Level 2 @112
    run_dt(XB, 112, 112, 1, 1, 12);
    run_conv(F2, 64, 18, 128, 112, 112, 1, 1, F1);
    run_pool(F1, 128, 112, 112, XB, F2, XA);
    // Level 3 @56
    run_dt(XA, 56, 56, 1, 1, 20);
    run_conv(F2, 128, 26, 256, 56, 56, 1, 1, F1);
    run_conv(F1, 256, 28, 256, 56, 56, 1, 1, F2);
    run_pool(F2, 256, 56, 56, XA, F1, XB);
    // Level 4 @28
    run_dt(XB, 28, 28, 1, 1, 30);
    run_conv(F1, 256, 36, 512, 28, 28, 1, 1, F2);
    run_conv(F2, 512, 38, 512, 28, 28, 1, 1, F1);
    run_pool(F1, 512, 28, 28, XB, F2, XA);
    // Level 5 @14, dilation 2
    run_dt(XA, 14, 14, 2, 2, 40);
    run_conv(F2, 512, 46, 512, 14, 14, 2, 2, F1);
    run_conv(F1, 512, 48, 512, 14, 14, 2, 2, F2);
    // pool5 (max s1 p1) -> pool5a (avg, count_include_pad)
    {
        size_t n = (size_t)B * 512 * 14 * 14;
        pool_s1_kernel<<<cdiv(n, 256), 256, 0, stream>>>(F2, F1, 512, 14, 14, 0, n);
        pool_s1_kernel<<<cdiv(n, 256), 256, 0, stream>>>(F1, F2, 512, 14, 14, 1, n);
    }
    // FC stack on [B*196, 512]
    float* FCX = SM;
    float* FCA = SM + (1 << 20);
    float* FCB = SM + (2 << 20);
    {
        size_t n = (size_t)B * 196 * 512;
        chw2rows_kernel<<<cdiv(n, 256), 256, 0, stream>>>(F2, FCX, 512, 196, n);
        int M = B * 196;
        dim3 g6(cdiv(M, 16), cdiv(1024, 64), 1);
        gemm_xwT_wmma_kernel<<<g6, 128, 0, stream>>>(FCX, P(50), P(51), FCA, M, 512, 1024, 1);
        dim3 g7(cdiv(M, 16), cdiv(1024, 64), 1);
        gemm_xwT_wmma_kernel<<<g7, 128, 0, stream>>>(FCA, P(52), P(53), FCB, M, 1024, 1024, 1);
        dim3 g8(cdiv(M, 16), cdiv(40, 64), 1);
        gemm_xwT_wmma_kernel<<<g8, 128, 0, stream>>>(FCB, P(54), P(55), (float*)d_out,
                                                    M, 1024, 40, 0);
    }
}
